// NeuralAstar_76029511074519
// MI455X (gfx1250) — matrix operand order repaired and checked by reading, not compiled, not run on hardware
//
#include <hip/hip_runtime.h>
#include <hip/hip_bf16.h>

typedef __attribute__((ext_vector_type(16))) _Float16 v16h;
typedef __attribute__((ext_vector_type(8)))  float    v8f;

#define NB    64
#define HH    32
#define WW    32
#define NPIX  1024   // 32*32
#define STEPS 256    // int(0.25*32*32)

// ---------------------------------------------------------------------------
// Kernel 1: encoder.  One block per batch element, 256 threads (8 waves).
// conv1 (2->32ch, 3x3, SAME) as implicit GEMM via v_wmma_f32_16x16x32_f16:
//   M=32 (out ch, 2 tiles) x K=18->pad32 (im2col) x N=1024 (pixels, 64 tiles)
// mt is an unrolled outer loop so each WMMA has a fixed A fragment (no
// per-iteration cndmask selection of A registers).
// conv2 (32->1ch) + sigmoid as direct VALU dot product.
// ---------------------------------------------------------------------------
__global__ __launch_bounds__(256) void na_encoder_kernel(
    const float* __restrict__ maps,   // (64,1,32,32)
    const float* __restrict__ start,  // (64,1,32,32)
    const float* __restrict__ goal,   // (64,1,32,32)
    const float* __restrict__ w1,     // (32,2,3,3)
    const float* __restrict__ b1,     // (32,)
    const float* __restrict__ w2,     // (1,32,3,3)
    const float* __restrict__ b2,     // (1,)
    float* __restrict__ cost_out)     // (64,1,32,32)
{
  extern __shared__ float smem[];
  float*     in0 = smem;             // 1024 : map_designs[b]
  float*     in1 = in0 + NPIX;       // 1024 : start+goal
  float*     w2s = in1 + NPIX;       // 288
  float*     b1s = w2s + 288;        // 32
  _Float16*  Wh  = (_Float16*)(b1s + 32);          // 32x32 f16 (K padded 18->32)
  _Float16*  hid = (_Float16*)(Wh + 32 * 32);      // 32 ch x 1024 pix, f16

  const int t = threadIdx.x;
  const int b = blockIdx.x;

  // ---- stage inputs & weights into LDS ----
  for (int p = t; p < NPIX; p += 256) {
    in0[p] = maps[b * NPIX + p];
    in1[p] = start[b * NPIX + p] + goal[b * NPIX + p];
  }
  for (int i = t; i < 32 * 32; i += 256) {
    int m = i >> 5, k = i & 31;
    float v = (k < 18) ? w1[m * 18 + k] : 0.0f;   // k = ci*9 + (dy+1)*3 + (dx+1)
    Wh[i] = (_Float16)v;
  }
  for (int i = t; i < 288; i += 256) w2s[i] = w2[i];
  if (t < 32) b1s[t] = b1[t];
  __syncthreads();

  const int lane = t & 31;
  const int wave = t >> 5;
  const int half = lane >> 4;    // 0: lanes 0-15, 1: lanes 16-31
  const int lr   = lane & 15;

  // ---- conv1: outer M-tile loop (fixed A per WMMA), 64 pixel tiles inner ----
#pragma unroll
  for (int mt = 0; mt < 2; ++mt) {
    // A fragment (weights), 16x32 f16 layout:
    // lane row M = mt*16 + lr ; i<8 -> K=i+8*half, i>=8 -> K=i+8+8*half
    v16h afrag;
    {
      int row = mt * 16 + lr;
#pragma unroll
      for (int i = 0; i < 16; ++i) {
        int k = (i < 8 ? i : i + 8) + 8 * half;
        afrag[i] = Wh[row * 32 + k];
      }
    }

    for (int pt = wave; pt < 64; pt += 8) {
      int pix = pt * 16 + lr;
      int y = pix >> 5, x = pix & 31;

      // B fragment (im2col patch), 32x16 f16 layout: i<8 -> K=i+8*half, i>=8 -> K=i+8+8*half, N = lr
      v16h bfrag;
#pragma unroll
      for (int i = 0; i < 16; ++i) {
        int k = (i < 8 ? i : i + 8) + 8 * half;
        float v = 0.0f;
        if (k < 18) {
          int ci = k / 9, r = k % 9;
          int yy = y + r / 3 - 1;
          int xx = x + r % 3 - 1;
          if (yy >= 0 && yy < HH && xx >= 0 && xx < WW)
            v = (ci ? in1 : in0)[yy * WW + xx];
        }
        bfrag[i] = (_Float16)v;
      }

      v8f c = {};
      c = __builtin_amdgcn_wmma_f32_16x16x32_f16(
          /*neg_a=*/false, afrag, /*neg_b=*/false, bfrag,
          /*c_mod=*/(short)0, c, /*reuse_a=*/false, /*reuse_b=*/false);

      // C layout: vgpr jj -> M = jj + 8*half, N = lr
#pragma unroll
      for (int jj = 0; jj < 8; ++jj) {
        int ch = mt * 16 + jj + 8 * half;
        float v = c[jj] + b1s[ch];
        hid[ch * NPIX + pix] = (_Float16)(v > 0.0f ? v : 0.0f);   // relu
      }
    }
  }
  __syncthreads();

  // ---- conv2 (32->1) + sigmoid ----
  const float bb = b2[0];
  for (int p = t; p < NPIX; p += 256) {
    int y = p >> 5, x = p & 31;
    float acc = bb;
    for (int ci = 0; ci < 32; ++ci) {
      const _Float16* hc = hid + ci * NPIX;
      const float*    wc = w2s + ci * 9;
#pragma unroll
      for (int r = 0; r < 9; ++r) {
        int yy = y + r / 3 - 1;
        int xx = x + r % 3 - 1;
        if (yy >= 0 && yy < HH && xx >= 0 && xx < WW)
          acc += wc[r] * (float)hc[yy * WW + xx];
      }
    }
    cost_out[b * NPIX + p] = 1.0f / (1.0f + __expf(-acc));   // sigmoid
  }
}

// ---------------------------------------------------------------------------
// Kernel 2: differentiable A* forward + backtrack.  One block per batch
// element, 256 threads (8 wave32s), all state in LDS.
//
// Forward value of st_softmax_noexp == one_hot(argmax(exp(-f/sqrt32)*open))
// == argmin f over open cells (tie -> lowest index, exp is monotone).
// Packed u64 (f_bits<<32 | idx) min is exact since f >= 0.
//
// Per step only TWO __syncthreads: intra-wave shuffle reduction (no LDS/
// barrier), one barrier, then wave 0 alone finishes the 8-value reduction
// with shuffles and applies the sparse 9-cell update (sel + 8 neighbors are
// disjoint cells; each neighbor thread recomputes g2 = g[sel]+cost[sel]
// locally), then the end-of-step barrier.
// ---------------------------------------------------------------------------
__global__ __launch_bounds__(256) void na_astar_kernel(
    const float* __restrict__ maps,     // obstacles (64,1,32,32)
    const float* __restrict__ start,    // (64,1,32,32)
    const float* __restrict__ goal,     // (64,1,32,32)
    const float* __restrict__ cost_in,  // pred_cost_maps (64,1,32,32)
    float* __restrict__ hist_out,       // (64,1,32,32)
    float* __restrict__ path_out)       // (64,1,32,32)
{
  __shared__ float cost[NPIX], hmap[NPIX], g[NPIX], openm[NPIX], hist[NPIX], obs[NPIX];
  __shared__ int   par[NPIX];
  __shared__ unsigned char pathb[NPIX];
  __shared__ unsigned long long red[8];
  __shared__ int   s_goal;

  const int t    = threadIdx.x;
  const int b    = blockIdx.x;
  const int lane = t & 31;
  const int wave = t >> 5;

  __builtin_prefetch(&cost_in[b * NPIX], 0, 0);   // global_prefetch_b8
  __builtin_prefetch(&maps[b * NPIX], 0, 0);

  if (t == 0) s_goal = 0;
  __syncthreads();

  for (int p = t; p < NPIX; p += 256) {
    float gv = goal[b * NPIX + p];
    if (gv > 0.5f) s_goal = p;                 // goal is one-hot: single writer
    cost[p]  = cost_in[b * NPIX + p];
    obs[p]   = maps[b * NPIX + p];
    g[p]     = 0.0f;
    openm[p] = start[b * NPIX + p];
    hist[p]  = 0.0f;
    pathb[p] = (gv > 0.5f) ? 1 : 0;            // path initialized to goal map
  }
  __syncthreads();

  const int gi = s_goal;
  const int gy = gi >> 5, gx = gi & 31;
  for (int p = t; p < NPIX; p += 256) {
    float dy = fabsf((float)((p >> 5) - gy));
    float dx = fabsf((float)((p & 31) - gx));
    // d.min + d.max == dy+dx ; + tb_factor * euclidean ; + cost
    hmap[p] = (dy + dx) + 0.001f * sqrtf(dy * dy + dx * dx) + cost[p];
    par[p] = gi;
  }
  __syncthreads();

  for (int step = 0; step < STEPS; ++step) {
    // --- packed argmin of f over open cells (4 cells per thread) ---
    unsigned long long best = ~0ull;
#pragma unroll
    for (int pp = 0; pp < 4; ++pp) {
      int p = t + pp * 256;
      float f = 0.5f * g[p] + 0.5f * hmap[p];
      unsigned fb = (openm[p] > 0.5f) ? __float_as_uint(f) : 0x7FFFFFFFu;
      unsigned long long key = ((unsigned long long)fb << 32) | (unsigned)p;
      best = (key < best) ? key : best;
    }
    // intra-wave min via lane shuffles (no barrier)
#pragma unroll
    for (int off = 16; off > 0; off >>= 1) {
      unsigned long long o = __shfl_xor(best, off, 32);
      best = (o < best) ? o : best;
    }
    if (lane == 0) red[wave] = best;
    __syncthreads();                                    // barrier #1

    if (wave == 0) {
      // fold the 8 per-wave results inside wave 0 (values replicated /8 lanes)
      unsigned long long v = red[lane & 7];
#pragma unroll
      for (int off = 4; off > 0; off >>= 1) {
        unsigned long long o = __shfl_xor(v, off, 32);
        v = (o < v) ? o : v;
      }
      const int sel = (int)(v & 0xFFFFFFFFull);

      if (lane < 8) {
        // 8-neighbor sparse update: expand(sel) is nonzero only there
        const int dyo[8] = {-1, -1, -1, 0, 0, 1, 1, 1};
        const int dxo[8] = {-1, 0, 1, -1, 1, -1, 0, 1};
        int ny = (sel >> 5) + dyo[lane];
        int nx = (sel & 31) + dxo[lane];
        if (ny >= 0 && ny < HH && nx >= 0 && nx < WW) {
          int p = ny * WW + nx;
          float nbr = obs[p];                           // expand(sel)*obs
          float op = openm[p], hi = hist[p];
          float g2 = g[sel] + cost[sel];                // expand((g+cost)*sel)
          float idx = ((1.0f - op) * (1.0f - hi) +
                       op * ((g[p] > g2) ? 1.0f : 0.0f)) * nbr;
          if (idx > 0.5f) {                             // idx in {0,1}
            g[p] = g2;                                  // g = g2*idx + g*(1-idx)
            openm[p] = 1.0f;                            // clip(open+idx,0,1)
            par[p] = sel;                               // parents update
          }
        }
      } else if (lane == 8) {
        // scalar updates at sel (disjoint from all 8 neighbor cells)
        hist[sel] = 1.0f;                               // clip(hist+sel,0,1)
        if (sel != gi)                                  // is_unsolved
          openm[sel] = fmaxf(openm[sel] - 1.0f, 0.0f);  // clip(open-sel,0,1)
      }
    }
    __syncthreads();                                    // barrier #2
  }

  // --- backtrack: current_t = STEPS-1 iterations from parents[goal] ---
  if (t == 0) {
    int loc = par[gi];
    for (int i = 0; i < STEPS - 1; ++i) {
      pathb[loc] = 1;
      loc = par[loc];
    }
  }
  __syncthreads();

  for (int p = t; p < NPIX; p += 256) {
    hist_out[b * NPIX + p] = hist[p];
    path_out[b * NPIX + p] = (float)pathb[p];
  }
}

// ---------------------------------------------------------------------------
// d_in order: map_designs, start_maps, goal_maps, w1, b1, w2, b2
// d_out: [histories (65536) | paths (65536) | pred_cost_maps (65536)]
// ---------------------------------------------------------------------------
extern "C" void kernel_launch(void* const* d_in, const int* in_sizes, int n_in,
                              void* d_out, int out_size, void* d_ws, size_t ws_size,
                              hipStream_t stream) {
  const float* maps  = (const float*)d_in[0];
  const float* start = (const float*)d_in[1];
  const float* goal  = (const float*)d_in[2];
  const float* w1    = (const float*)d_in[3];
  const float* b1    = (const float*)d_in[4];
  const float* w2    = (const float*)d_in[5];
  const float* b2    = (const float*)d_in[6];

  float* out      = (float*)d_out;
  float* hist_out = out;
  float* path_out = out + NB * NPIX;
  float* cost_out = out + 2 * NB * NPIX;

  // LDS: in0+in1 (2*1024 f32) + w2s(288) + b1s(32) f32, Wh (32*32 f16),
  //      hidden (32*1024 f16)
  size_t smem1 = (size_t)(2 * NPIX + 288 + 32) * sizeof(float)
               + (size_t)(32 * 32 + 32 * NPIX) * sizeof(_Float16);

  na_encoder_kernel<<<NB, 256, smem1, stream>>>(maps, start, goal, w1, b1, w2,
                                                b2, cost_out);
  na_astar_kernel<<<NB, 256, 0, stream>>>(maps, start, goal, cost_out,
                                          hist_out, path_out);
}
